// MutiHeadAttention_27848567948122
// MI455X (gfx1250) — matrix operand order repaired and checked by reading, not compiled, not run on hardware
//
#include <hip/hip_runtime.h>
#include <hip/hip_bf16.h>
#include <math.h>

#define D_MODEL 1024
#define NHEAD   16
#define D_HEAD  64
#define BATCH   2
#define LSEQ    2048
#define MROWS   (BATCH*LSEQ)   // 4096

// Async global->LDS staging (GLOBAL_LOAD_ASYNC_TO_LDS_B128 + s_wait_asynccnt),
// guarded so a toolchain without the builtins falls back to sync copies.
#if defined(__has_builtin)
#if __has_builtin(__builtin_amdgcn_global_load_async_to_lds_b128) && \
    __has_builtin(__builtin_amdgcn_s_wait_asynccnt)
#define USE_ASYNC_LDS 1
#endif
#endif
#ifndef USE_ASYNC_LDS
#define USE_ASYNC_LDS 0
#endif

typedef __attribute__((ext_vector_type(16))) __bf16 v16bf;
typedef __attribute__((ext_vector_type(8)))  __bf16 v8bf;
typedef __attribute__((ext_vector_type(8)))  float  v8f;
// exact pointee type of the async-load builtin's pointer params (per clang
// diagnostic: "__attribute__((__vector_size__(4 * sizeof(int)))) int")
typedef __attribute__((__vector_size__(4 * sizeof(int)))) int v4i_async;

__device__ __forceinline__ v8bf cvt8(const float4* p) {
  float4 x = p[0], y = p[1];
  v8bf r;
  r[0] = (__bf16)x.x; r[1] = (__bf16)x.y; r[2] = (__bf16)x.z; r[3] = (__bf16)x.w;
  r[4] = (__bf16)y.x; r[5] = (__bf16)y.y; r[6] = (__bf16)y.z; r[7] = (__bf16)y.w;
  return r;
}

__device__ __forceinline__ v16bf join16(v8bf lo, v8bf hi) {
  return __builtin_shufflevector(lo, hi, 0,1,2,3,4,5,6,7,8,9,10,11,12,13,14,15);
}

__device__ __forceinline__ v8f wmma_bf16(v16bf a, v16bf b, v8f c) {
  // D(16x16,f32) = A(16x32,bf16) * B(32x16,bf16) + C
  return __builtin_amdgcn_wmma_f32_16x16x32_bf16(false, a, false, b, (short)0, c,
                                                 false, false);
}

// Raw v_exp_f32: args are always <= 0 here (post max-subtraction), so we do
// not need libm's overflow clamp.
__device__ __forceinline__ float fast_exp2(float x) {
#if defined(__has_builtin) && __has_builtin(__builtin_amdgcn_exp2f)
  return __builtin_amdgcn_exp2f(x);
#else
  return exp2f(x);
#endif
}

// one 16B global->LDS transfer per calling lane
__device__ __forceinline__ void copy16_to_lds(const __bf16* g, __bf16* l) {
#if USE_ASYNC_LDS
  __builtin_amdgcn_global_load_async_to_lds_b128(
      (__attribute__((address_space(1))) v4i_async*)g,
      (__attribute__((address_space(3))) v4i_async*)l, 0, 0);
#else
  *(v8bf*)l = *(const v8bf*)g;
#endif
}

__device__ __forceinline__ void wait_async_copies() {
#if USE_ASYNC_LDS
  __builtin_amdgcn_s_wait_asynccnt(0);
#endif
}

// ---------------------------------------------------------------------------
// GEMM: D = X @ W^T + bias.  X: [MROWS, D_MODEL] (fp32 or bf16, XBF16 flag),
// W: [D_MODEL, D_MODEL] fp32 row-major (W^T column n == W row n, so both
// operands walk K contiguously -> direct WMMA fragment loads; entire working
// set is L2-resident on MI455X's 192MB L2).
// Each wave computes a 32x64 output tile: 2 m-tiles x 4 n-tiles, 8 WMMAs/chunk.
// mode 0: fp32 row-major [MROWS, D_MODEL]
// mode 1: bf16 scatter to [B, H, L, Dh]   (q / k for attention)
// mode 2: bf16 scatter to [B, H, Dh, L]   (v transposed for attention)
// ---------------------------------------------------------------------------
template<int XBF16>
__global__ __launch_bounds__(256) void gemm_xwT_bias(
    const void* __restrict__ Xv, const float* __restrict__ W,
    const float* __restrict__ bias, void* __restrict__ out, int mode)
{
  const int tid   = blockIdx.x * blockDim.x + threadIdx.x;
  const int wid   = tid >> 5;
  const int lane  = tid & 31;
  const int lhalf = lane >> 4;
  const int lmod  = lane & 15;
  const int nblk  = wid & 15;      // D_MODEL / 64 = 16 n-blocks
  const int mblk  = wid >> 4;
  const int mbase = mblk * 32;
  const int nbase = nblk * 64;
  if (mbase >= MROWS) return;

  v8f c[2][4];
#pragma unroll
  for (int nt = 0; nt < 4; ++nt) {
    float bv = bias[nbase + nt * 16 + lmod];   // C-frag column is lane%16
#pragma unroll
    for (int mt = 0; mt < 2; ++mt)
#pragma unroll
      for (int i = 0; i < 8; ++i) c[mt][nt][i] = bv;
  }

  for (int k0 = 0; k0 < D_MODEL; k0 += 32) {
    v16bf a[2];
#pragma unroll
    for (int mt = 0; mt < 2; ++mt) {
      // A frag (16x32): lane m holds X[m][k0+8h+0..7] and X[m][k0+16+8h+0..7]
      const int roff = (mbase + mt * 16 + lmod) * D_MODEL + k0;
      if (XBF16) {
        const __bf16* rp = (const __bf16*)Xv + roff;
        a[mt] = join16(*(const v8bf*)(rp + 8 * lhalf),
                       *(const v8bf*)(rp + 16 + 8 * lhalf));
      } else {
        const float* rp = (const float*)Xv + roff;
        a[mt] = join16(cvt8((const float4*)(rp + 8 * lhalf)),
                       cvt8((const float4*)(rp + 16 + 8 * lhalf)));
      }
    }
#pragma unroll
    for (int nt = 0; nt < 4; ++nt) {
      // B frag (32x16): lane n holds W[n][k0+8h+0..7] and W[n][k0+16+8h+0..7]
      const float* wp = W + (nbase + nt * 16 + lmod) * D_MODEL + k0 + 8 * lhalf;
      v16bf bf = join16(cvt8((const float4*)wp), cvt8((const float4*)(wp + 16)));
#pragma unroll
      for (int mt = 0; mt < 2; ++mt)
        c[mt][nt] = wmma_bf16(a[mt], bf, c[mt][nt]);
    }
  }

#pragma unroll
  for (int mt = 0; mt < 2; ++mt)
#pragma unroll
    for (int nt = 0; nt < 4; ++nt)
#pragma unroll
      for (int v = 0; v < 8; ++v) {
        int row = mbase + mt * 16 + 8 * lhalf + v;  // C layout: row = 8*half+v
        int col = nbase + nt * 16 + lmod;           // col = lane%16
        float val = c[mt][nt][v];
        if (mode == 0) {
          ((float*)out)[row * D_MODEL + col] = val;
        } else {
          int b_ = row >> 11, l = row & (LSEQ - 1);
          int h  = col >> 6,  d = col & 63;
          __bf16* o = (__bf16*)out;
          if (mode == 1)
            o[((b_ * NHEAD + h) * LSEQ + l) * D_HEAD + d] = (__bf16)val;
          else
            o[((b_ * NHEAD + h) * D_HEAD + d) * LSEQ + l] = (__bf16)val;
        }
      }
}

// ---------------------------------------------------------------------------
// Flash attention. One block = 8 waves, all on the same (b, h); wave w owns
// query block qgroup*8+w. K (32x64 bf16) and V^T (64x32 bf16) chunks are
// double-buffered in LDS: the next chunk is staged with async global->LDS
// b128 transfers (ASYNCcnt) while the current chunk's 8 WMMAs execute, then
// s_wait_asynccnt 0 + workgroup barrier. 8 waves share each staged tile ->
// 8x less L2 read traffic, and global latency hides behind the matrix math.
// Computes S^T = K_chunk @ Q^T so the exp'd C-fragment of S^T is bit-exact
// the A-fragment layout for P @ V (no cross-lane transpose).
// ---------------------------------------------------------------------------
__global__ __launch_bounds__(256) void flash_attn_fwd(
    const __bf16* __restrict__ qg, const __bf16* __restrict__ kg,
    const __bf16* __restrict__ vtg, __bf16* __restrict__ aout)
{
  __shared__ __align__(128) __bf16 sk[2][32 * 64];  // K tiles  [key][d]
  __shared__ __align__(128) __bf16 sv[2][64 * 32];  // V^T tiles [d][key]

  const int t      = threadIdx.x;
  const int wave   = t >> 5;
  const int lane   = t & 31;
  const int lhalf  = lane >> 4;
  const int lmod   = lane & 15;
  const int bh     = blockIdx.x >> 4;      // 0 .. B*H-1
  const int qgroup = blockIdx.x & 15;
  const int qblk   = qgroup * 8 + wave;    // 128 query blocks per (b,h)

  const __bf16* qp = qg  + (bh * LSEQ + qblk * 16) * D_HEAD;
  const __bf16* kp = kg  + bh * LSEQ * D_HEAD;
  const __bf16* vp = vtg + bh * D_HEAD * LSEQ;

  // cooperative-staging indices: one 16B b128 transfer per thread per tile
  const int krow = t >> 3, kcol = (t & 7) * 8;   // sk: 32 rows x 128B
  const int vrow = t >> 2, vcol = (t & 3) * 8;   // sv: 64 rows x 64B

  // Q as B-fragments (d-chunk g): lane n = query, d = g*32+8h+0..7 and g*32+16+8h+0..7
  v16bf qf[2];
#pragma unroll
  for (int g = 0; g < 2; ++g)
    qf[g] = join16(*(const v8bf*)(qp + lmod * D_HEAD + g * 32 + 8 * lhalf),
                   *(const v8bf*)(qp + lmod * D_HEAD + g * 32 + 16 + 8 * lhalf));

  v8f o[4];
#pragma unroll
  for (int dt = 0; dt < 4; ++dt)
#pragma unroll
    for (int i = 0; i < 8; ++i) o[dt][i] = 0.0f;

  float mrun = -1e30f, lrun = 0.0f;
  const float SC = 0.125f * 1.4426950408889634f;  // 1/sqrt(Dh) * log2(e)
  const int sb = lhalf << 3;                       // broadcast source base lane

  // stage chunk 0
  copy16_to_lds(kp + krow * D_HEAD + kcol, &sk[0][krow * 64 + kcol]);
  copy16_to_lds(vp + vrow * LSEQ + vcol,   &sv[0][vrow * 32 + vcol]);
  wait_async_copies();
  __syncthreads();

  for (int j0 = 0; j0 < LSEQ; j0 += 32) {
    const int cur = (j0 >> 5) & 1, nxt = cur ^ 1;
    if (j0 + 32 < LSEQ) {  // async-stage the next chunk behind the WMMAs
      copy16_to_lds(kp + (j0 + 32 + krow) * D_HEAD + kcol,
                    &sk[nxt][krow * 64 + kcol]);
      copy16_to_lds(vp + vrow * LSEQ + (j0 + 32) + vcol,
                    &sv[nxt][vrow * 32 + vcol]);
    }

    // S^T (keys x queries) for the two 16-key groups, K over Dh = 2 chunks
    v8f s[2];
#pragma unroll
    for (int grp = 0; grp < 2; ++grp) {
      v8f acc;
#pragma unroll
      for (int i = 0; i < 8; ++i) acc[i] = 0.0f;
      const __bf16* rp = &sk[cur][(grp * 16 + lmod) * 64];
#pragma unroll
      for (int g = 0; g < 2; ++g) {
        v16bf ka = join16(*(const v8bf*)(rp + g * 32 + 8 * lhalf),
                          *(const v8bf*)(rp + g * 32 + 16 + 8 * lhalf));
        acc = wmma_bf16(ka, qf[g], acc);
      }
      s[grp] = acc;
    }

    // online softmax: this lane owns query lmod, keys {8h+v, 16+8h+v}
    float tv[16];
#pragma unroll
    for (int i = 0; i < 8; ++i) { tv[i] = s[0][i] * SC; tv[8 + i] = s[1][i] * SC; }
    float cm = tv[0];
#pragma unroll
    for (int i = 1; i < 16; ++i) cm = fmaxf(cm, tv[i]);
    cm = fmaxf(cm, __shfl_xor(cm, 16, 32));
    float mnew  = fmaxf(mrun, cm);
    float alpha = fast_exp2(mrun - mnew);
    v16bf pa;
    float sum = 0.0f;
#pragma unroll
    for (int i = 0; i < 16; ++i) {
      float p = fast_exp2(tv[i] - mnew);
      sum += p;
      pa[i] = (__bf16)p;   // exp'd S^T C-frag IS the P A-frag: no shuffles
    }
    sum += __shfl_xor(sum, 16, 32);
    lrun = lrun * alpha + sum;
    mrun = mnew;

    // rescale O: rows of the C-frag are queries 8h+v -> broadcast alpha
    float scl[8];
#pragma unroll
    for (int v = 0; v < 8; ++v) scl[v] = __shfl(alpha, sb + v, 32);
#pragma unroll
    for (int dt = 0; dt < 4; ++dt)
#pragma unroll
      for (int v = 0; v < 8; ++v) o[dt][v] *= scl[v];

    // O += P @ V : B-frag = keys 8h+0..7 and 16+8h+0..7 from the LDS V^T tile
#pragma unroll
    for (int dt = 0; dt < 4; ++dt) {
      const __bf16* vr = &sv[cur][(dt * 16 + lmod) * 32 + 8 * lhalf];
      o[dt] = wmma_bf16(pa, join16(*(const v8bf*)vr, *(const v8bf*)(vr + 16)), o[dt]);
    }

    wait_async_copies();   // my next-chunk transfers have landed
    __syncthreads();       // everyone's have; cur buffer free for reuse
  }

  // normalize and store heads-concatenated bf16 [B*L, D_MODEL]
  float linv = 1.0f / lrun;
  float lv[8];
#pragma unroll
  for (int v = 0; v < 8; ++v) lv[v] = __shfl(linv, sb + v, 32);
  const int b_ = bh >> 4, h = bh & 15;
#pragma unroll
  for (int dt = 0; dt < 4; ++dt)
#pragma unroll
    for (int v = 0; v < 8; ++v) {
      int row = qblk * 16 + 8 * lhalf + v;
      int addr = (b_ * LSEQ + row) * D_MODEL + h * D_HEAD + dt * 16 + lmod;
      aout[addr] = (__bf16)(o[dt][v] * lv[v]);
    }
}

extern "C" void kernel_launch(void* const* d_in, const int* in_sizes, int n_in,
                              void* d_out, int out_size, void* d_ws, size_t ws_size,
                              hipStream_t stream) {
  const float* Q  = (const float*)d_in[0];
  const float* K  = (const float*)d_in[1];
  const float* V  = (const float*)d_in[2];
  const float* Wq = (const float*)d_in[3];
  const float* bq = (const float*)d_in[4];
  const float* Wk = (const float*)d_in[5];
  const float* bk = (const float*)d_in[6];
  const float* Wv = (const float*)d_in[7];
  const float* bv = (const float*)d_in[8];
  const float* Wo = (const float*)d_in[9];
  const float* bo = (const float*)d_in[10];

  const size_t SEG = (size_t)MROWS * D_MODEL * sizeof(__bf16);  // 8 MB
  char* ws = (char*)d_ws;
  __bf16* qb = (__bf16*)(ws);            // [B,H,L,Dh]
  __bf16* kb = (__bf16*)(ws + SEG);      // [B,H,L,Dh]
  __bf16* vb = (__bf16*)(ws + 2 * SEG);  // [B,H,Dh,L]
  __bf16* ab = (__bf16*)(ws + 3 * SEG);  // [B*L, D_MODEL]

  // (MROWS/32)*(D_MODEL/64) = 2048 waves, 8 waves/block -> 256 blocks
  dim3 gblk(256), gthr(256);
  gemm_xwT_bias<0><<<gblk, gthr, 0, stream>>>((const void*)Q, Wq, bq, (void*)qb, 1);
  gemm_xwT_bias<0><<<gblk, gthr, 0, stream>>>((const void*)K, Wk, bk, (void*)kb, 1);
  gemm_xwT_bias<0><<<gblk, gthr, 0, stream>>>((const void*)V, Wv, bv, (void*)vb, 2);

  // 32 (b,h) x 16 qgroups = 512 blocks of 8 waves
  flash_attn_fwd<<<dim3(512), dim3(256), 0, stream>>>(qb, kb, vb, ab);

  gemm_xwT_bias<1><<<gblk, gthr, 0, stream>>>((const void*)ab, Wo, bo, d_out, 0);
}
